// StandaloneBayesianNeuralDistance_29291676959063
// MI455X (gfx1250) — matrix operand order repaired and checked by reading, not compiled, not run on hardware
//
#include <hip/hip_runtime.h>

#define D   512
#define NQ  4096
#define NP  1024
#define NS  10
#define BN  64            // block N-tile (rows of the staged B panel)
#define PADK (D + 8)      // padded LDS row stride (elements) -> breaks bank period

typedef __attribute__((ext_vector_type(16))) __bf16 v16bf;
typedef __attribute__((ext_vector_type(8)))  __bf16 v8bf;
typedef __attribute__((ext_vector_type(8)))  float  v8f;

// ---------------- deterministic "gaussian" noise (stands in for jax PRNG) ----
__device__ __forceinline__ unsigned pcg_hash(unsigned x) {
    x = x * 747796405u + 2891336453u;
    x = ((x >> ((x >> 28) + 4u)) ^ x) * 277803737u;
    return (x >> 22) ^ x;
}
__device__ __forceinline__ float hash_gauss(unsigned idx, unsigned salt) {
    unsigned h1 = pcg_hash(idx ^ salt);
    unsigned h2 = pcg_hash(h1 + 0x632BE5ABu);
    float u1 = ((float)(h1 >> 8) + 0.5f) * (1.0f / 16777216.0f);
    float u2 = ((float)(h2 >> 8) + 0.5f) * (1.0f / 16777216.0f);
    return sqrtf(-2.0f * logf(u1)) * cosf(6.28318530718f * u2);
}

// ---- stage a [BN x D] bf16 panel (rows n0..n0+BN-1 of src) into padded LDS --
// Uses CDNA5 async global->LDS DMA (ASYNCcnt), then fences + barriers.
// 64 rows * 64 chunks(16B) = 4096 chunks; 256 threads -> 16 issues/thread.
__device__ __forceinline__ void stage_panel_async(const __bf16* __restrict__ src,
                                                  int n0, __bf16* lds_panel) {
    const int tid = threadIdx.x;
#pragma unroll
    for (int p = 0; p < 16; ++p) {
        int c   = p * 256 + tid;
        int row = c >> 6;          // 0..63
        int cin = c & 63;          // 16B chunk within row
        const __bf16* g = src + (size_t)(n0 + row) * D + cin * 8;
        unsigned lds = (unsigned)(size_t)(lds_panel + row * PADK + cin * 8);
        asm volatile("global_load_async_to_lds_b128 %0, %1, off"
                     :: "v"(lds), "v"(g) : "memory");
    }
    asm volatile("s_wait_asynccnt 0x0" ::: "memory");
    __syncthreads();
}

// ---------------- prep kernels ----------------------------------------------
__global__ void k_cast_q(const float* __restrict__ q, __bf16* __restrict__ qbf) {
    int i = blockIdx.x * 256 + threadIdx.x;
    qbf[i] = (__bf16)q[i];
}

__global__ void k_prep_p(const float* __restrict__ p, __bf16* __restrict__ pbf,
                         float* __restrict__ sqp) {
    __shared__ float red[256];
    int row = blockIdx.x, t = threadIdx.x;
    float a = p[row * D + t], b = p[row * D + t + 256];
    pbf[row * D + t]       = (__bf16)a;
    pbf[row * D + t + 256] = (__bf16)b;
    red[t] = a * a + b * b;
    __syncthreads();
    for (int s = 128; s > 0; s >>= 1) {
        if (t < s) red[t] += red[t + s];
        __syncthreads();
    }
    if (t == 0) sqp[row] = red[0];
}

// Wt[s][e][k] = w[s][k][e] = mu[k][e] + softplus(rho[k][e]) * eps(s,k,e)
__global__ void k_gen_w(const float* __restrict__ mu, const float* __restrict__ rho,
                        __bf16* __restrict__ Wt) {
    unsigned i = blockIdx.x * 256u + threadIdx.x;      // i = ((s*D)+e)*D + k
    unsigned k = i % D;
    unsigned e = (i / D) % D;
    unsigned s = i / (D * D);
    float m  = mu[k * D + e];
    float sg = log1pf(expf(rho[k * D + e]));
    float ep = hash_gauss((s * D + k) * D + e, 0x9E3779B9u);
    Wt[i] = (__bf16)(m + sg * ep);
}

__global__ void k_gen_b(const float* __restrict__ bmu, const float* __restrict__ brho,
                        float* __restrict__ bs) {
    int i = blockIdx.x * 256 + threadIdx.x;
    if (i >= NS * D) return;
    int e = i % D;
    float sg = log1pf(expf(brho[e]));
    bs[i] = bmu[e] + sg * hash_gauss((unsigned)i, 0x7F4A7C15u);
}

// ---------------- GEMM1: tq = Q @ W_s + b_s  (f32 out) ----------------------
__global__ __launch_bounds__(256) void k_gemm_tq(
    const __bf16* __restrict__ qbf,    // [NQ][D]
    const __bf16* __restrict__ Wt_s,   // [D(e)][D(k)]  (= W_s transposed)
    const float*  __restrict__ bias_s, // [D]
    float* __restrict__ tqf)           // [NQ][D]
{
    __shared__ __bf16 Bt[BN * PADK];   // 66.5 KB padded panel

    const int lane  = threadIdx.x & 31;
    const int wave  = threadIdx.x >> 5;
    const int lan16 = lane & 15;
    const int half  = lane >> 4;
    const int m0 = (blockIdx.x * 8 + wave) * 16;
    const int n0 = blockIdx.y * BN;

    stage_panel_async(Wt_s, n0, Bt);   // async DMA of full K panel, once

    const v8f vzero = {0, 0, 0, 0, 0, 0, 0, 0};
    v8f acc[4];
#pragma unroll
    for (int j = 0; j < 4; ++j) acc[j] = vzero;

    const __bf16* arow = qbf + (size_t)(m0 + lan16) * D;
    for (int k0 = 0; k0 < D; k0 += 32) {
        const int kb = k0 + half * 16;
        v8bf lo = *(const v8bf*)(arow + kb);
        v8bf hi = *(const v8bf*)(arow + kb + 8);
        v16bf av = __builtin_shufflevector(lo, hi, 0, 1, 2, 3, 4, 5, 6, 7,
                                           8, 9, 10, 11, 12, 13, 14, 15);
        const int kbb = k0 + half * 16;
#pragma unroll
        for (int j = 0; j < 4; ++j) {
            v16bf bv = *(const v16bf*)(Bt + (j * 16 + lan16) * PADK + kbb);
            acc[j] = __builtin_amdgcn_wmma_f32_16x16x32_bf16(
                false, av, false, bv, (short)0, acc[j], false, false);
        }
    }
#pragma unroll
    for (int j = 0; j < 4; ++j) {
        int n = n0 + j * 16 + lan16;
        float bb = bias_s[n];
#pragma unroll
        for (int r = 0; r < 8; ++r) {
            int m = m0 + r + half * 8;
            tqf[(size_t)m * D + n] = acc[j][r] + bb;
        }
    }
}

// per-row: bf16 cast of tq + squared-norm
__global__ void k_tq_post(const float* __restrict__ tqf, __bf16* __restrict__ tqbf_s,
                          float* __restrict__ sqq_s) {
    __shared__ float red[256];
    int row = blockIdx.x, t = threadIdx.x;
    float a = tqf[row * D + t], b = tqf[row * D + t + 256];
    tqbf_s[row * D + t]       = (__bf16)a;
    tqbf_s[row * D + t + 256] = (__bf16)b;
    red[t] = a * a + b * b;
    __syncthreads();
    for (int s = 128; s > 0; s >>= 1) {
        if (t < s) red[t] += red[t + s];
        __syncthreads();
    }
    if (t == 0) sqq_s[row] = red[0];
}

// ---------------- GEMM2 fused over samples: dist stats -> mean/std ----------
__global__ __launch_bounds__(256) void k_dist_stats(
    const __bf16* __restrict__ tqbf,  // [NS][NQ][D]
    const __bf16* __restrict__ pbf,   // [NP][D]
    const float*  __restrict__ sqq,   // [NS][NQ]
    const float*  __restrict__ sqp,   // [NP]
    float* __restrict__ out)          // mean [NQ*NP], then std [NQ*NP]
{
    __shared__ __bf16 Bt[BN * PADK];  // prototype panel: shared by all 10 samples

    const int lane  = threadIdx.x & 31;
    const int wave  = threadIdx.x >> 5;
    const int lan16 = lane & 15;
    const int half  = lane >> 4;
    const int m0 = (blockIdx.x * 8 + wave) * 16;
    const int n0 = blockIdx.y * BN;

    stage_panel_async(pbf, n0, Bt);   // async DMA once; reused by all samples

    const v8f vzero = {0, 0, 0, 0, 0, 0, 0, 0};
    v8f sumd[4], sumdd[4];
#pragma unroll
    for (int j = 0; j < 4; ++j) { sumd[j] = vzero; sumdd[j] = vzero; }

    float sp[4];
#pragma unroll
    for (int j = 0; j < 4; ++j) sp[j] = sqp[n0 + j * 16 + lan16];

    for (int s = 0; s < NS; ++s) {
        const __bf16* arow = tqbf + (size_t)s * NQ * D + (size_t)(m0 + lan16) * D;
        v8f acc[4];
#pragma unroll
        for (int j = 0; j < 4; ++j) acc[j] = vzero;

        for (int k0 = 0; k0 < D; k0 += 32) {
            const int kb = k0 + half * 16;
            v8bf lo = *(const v8bf*)(arow + kb);
            v8bf hi = *(const v8bf*)(arow + kb + 8);
            v16bf av = __builtin_shufflevector(lo, hi, 0, 1, 2, 3, 4, 5, 6, 7,
                                               8, 9, 10, 11, 12, 13, 14, 15);
            const int kbb = k0 + half * 16;
#pragma unroll
            for (int j = 0; j < 4; ++j) {
                v16bf bv = *(const v16bf*)(Bt + (j * 16 + lan16) * PADK + kbb);
                acc[j] = __builtin_amdgcn_wmma_f32_16x16x32_bf16(
                    false, av, false, bv, (short)0, acc[j], false, false);
            }
        }
#pragma unroll
        for (int j = 0; j < 4; ++j) {
#pragma unroll
            for (int r = 0; r < 8; ++r) {
                int m = m0 + r + half * 8;
                float sqd  = sqq[s * NQ + m] + sp[j] - 2.0f * acc[j][r];
                float dist = sqrtf(fmaxf(sqd, 1e-12f));
                sumd[j][r]  += dist;
                sumdd[j][r] += dist * dist;
            }
        }
    }
#pragma unroll
    for (int j = 0; j < 4; ++j) {
        int n = n0 + j * 16 + lan16;
#pragma unroll
        for (int r = 0; r < 8; ++r) {
            int m = m0 + r + half * 8;
            float mean = sumd[j][r] * (1.0f / NS);
            float var  = (sumdd[j][r] - sumd[j][r] * sumd[j][r] * (1.0f / NS)) *
                         (1.0f / (NS - 1));
            size_t idx = (size_t)m * NP + n;
            out[idx]                    = mean;
            out[(size_t)NQ * NP + idx]  = sqrtf(fmaxf(var, 0.0f));
        }
    }
}

// ---------------- launcher ---------------------------------------------------
extern "C" void kernel_launch(void* const* d_in, const int* in_sizes, int n_in,
                              void* d_out, int out_size, void* d_ws, size_t ws_size,
                              hipStream_t stream) {
    const float* qf   = (const float*)d_in[0];
    const float* prot = (const float*)d_in[1];
    const float* wmu  = (const float*)d_in[2];
    const float* wrho = (const float*)d_in[3];
    const float* bmu  = (const float*)d_in[4];
    const float* brho = (const float*)d_in[5];
    float* out = (float*)d_out;

    char* ws = (char*)d_ws;
    size_t off = 0;
    auto alloc = [&](size_t bytes) -> char* {
        off = (off + 255) & ~(size_t)255;
        char* p = ws + off;
        off += bytes;
        return p;
    };
    __bf16* qbf  = (__bf16*)alloc((size_t)NQ * D * 2);
    __bf16* pbf  = (__bf16*)alloc((size_t)NP * D * 2);
    float*  sqp  = (float*) alloc((size_t)NP * 4);
    __bf16* Wt   = (__bf16*)alloc((size_t)NS * D * D * 2);
    float*  bs   = (float*) alloc((size_t)NS * D * 4);
    float*  tqf  = (float*) alloc((size_t)NQ * D * 4);
    __bf16* tqbf = (__bf16*)alloc((size_t)NS * NQ * D * 2);
    float*  sqq  = (float*) alloc((size_t)NS * NQ * 4);

    k_cast_q<<<(NQ * D) / 256, 256, 0, stream>>>(qf, qbf);
    k_prep_p<<<NP, 256, 0, stream>>>(prot, pbf, sqp);
    k_gen_w<<<(NS * D * D) / 256, 256, 0, stream>>>(wmu, wrho, Wt);
    k_gen_b<<<(NS * D + 255) / 256, 256, 0, stream>>>(bmu, brho, bs);

    for (int s = 0; s < NS; ++s) {
        k_gemm_tq<<<dim3(NQ / 128, D / 64), 256, 0, stream>>>(
            qbf, Wt + (size_t)s * D * D, bs + (size_t)s * D, tqf);
        k_tq_post<<<NQ, 256, 0, stream>>>(tqf, tqbf + (size_t)s * NQ * D,
                                          sqq + (size_t)s * NQ);
    }

    k_dist_stats<<<dim3(NQ / 128, NP / 64), 256, 0, stream>>>(tqbf, pbf, sqq, sqp, out);
}
